// LSTMFourierDecoder_82617990906569
// MI455X (gfx1250) — matrix operand order repaired and checked by reading, not compiled, not run on hardware
//
#include <hip/hip_runtime.h>
#include <hip/hip_bf16.h>
#include <math.h>

// Problem constants (match reference)
#define BB 32
#define FF 512
#define TT 128
#define DD 512
#define OO 512

typedef __attribute__((ext_vector_type(16))) __bf16 v16bf;
typedef __attribute__((ext_vector_type(8)))  float  v8f;

// ---------------------------------------------------------------------------
// WMMA fragment loaders for v_wmma_f32_16x16x32_bf16 (wave32).
//
// A (16x32, 16-bit): lane L (L&15 = row M), lanes>=16 get K offset +16.
//   vector elems 0..15 -> K = koff+0..15 (contiguous 32B)
// B (32x16, 16-bit), stored transposed Bt[n][k]: lane L -> col N=L&15,
//   lanes>=16 get K offset +16; elems 0..15 -> K = koff+0..15 (contiguous 32B)
// C/D (16x16 f32): elem r -> row = r + (lane>=16 ? 8:0), col = lane&15.
// ---------------------------------------------------------------------------
__device__ __forceinline__ v16bf load_a_frag(const __bf16* __restrict__ A,
                                             int lda, int m0, int k0) {
  const int lane = threadIdx.x & 31;
  const __bf16* p = A + (size_t)(m0 + (lane & 15)) * lda + k0 + ((lane >> 4) << 4);
  v16bf a;
#pragma unroll
  for (int e = 0; e < 16; ++e) a[e] = p[e];
  return a;
}

__device__ __forceinline__ v16bf load_b_frag(const __bf16* __restrict__ Bt,
                                             int ldb, int n0, int k0) {
  const int lane = threadIdx.x & 31;
  const __bf16* p = Bt + (size_t)(n0 + (lane & 15)) * ldb + k0 + ((lane >> 4) << 4);
  v16bf b;
#pragma unroll
  for (int e = 0; e < 16; ++e) b[e] = p[e];
  return b;
}

__device__ __forceinline__ v8f wmma_bf16(v16bf a, v16bf b, v8f c) {
  return __builtin_amdgcn_wmma_f32_16x16x32_bf16(false, a, false, b,
                                                 (short)0, c, false, false);
}

// ---------------------------------------------------------------------------
// Packing kernels: fp32 -> bf16, weights transposed to Bt[n][k].
// ---------------------------------------------------------------------------
__global__ void k_pack_x(const float* __restrict__ x, __bf16* __restrict__ xbf) {
  // x [B,F,T] -> xbf[(b*T+t)*F + f]
  int idx = blockIdx.x * blockDim.x + threadIdx.x;
  if (idx >= BB * FF * TT) return;
  int t = idx % TT;
  int r = idx / TT;
  int f = r % FF;
  int b = r / FF;
  xbf[((size_t)b * TT + t) * FF + f] = (__bf16)x[idx];
}

__global__ void k_pack_wT(const float* __restrict__ W, __bf16* __restrict__ Wt,
                          int K, int N) {
  // W[K][N] fp32 -> Wt[N][K] bf16
  int idx = blockIdx.x * blockDim.x + threadIdx.x;
  if (idx >= K * N) return;
  int n = idx / K;
  int k = idx % K;
  Wt[idx] = (__bf16)W[(size_t)k * N + n];
}

__global__ void k_init_state(const float* __restrict__ init_h,
                             const float* __restrict__ init_c,
                             __bf16* __restrict__ hinit,
                             float* __restrict__ c_ws) {
  int idx = blockIdx.x * blockDim.x + threadIdx.x;
  if (idx >= BB * DD) return;
  int d = idx % DD;
  hinit[idx] = (__bf16)init_h[d];
  c_ws[idx]  = init_c[d];
}

// ---------------------------------------------------------------------------
// Fourier features + LayerNorm: z = LN(cos(xt @ fourier_w + b))
// One block = 16 rows x 512 cols.  8 waves, each wave -> 16x64 strip.
// K-loop double-buffered: next A/B fragments in flight during current WMMAs.
// ---------------------------------------------------------------------------
__global__ __launch_bounds__(256) void k_fourier_ln(
    const __bf16* __restrict__ xbf, const __bf16* __restrict__ fwT,
    const float* __restrict__ fb, const float* __restrict__ gamma,
    const float* __restrict__ beta, __bf16* __restrict__ zbf) {
  __shared__ float zs[16][DD];
  __shared__ float red_s[16][16];
  __shared__ float red_q[16][16];

  const int tid = threadIdx.x, lane = tid & 31, wave = tid >> 5;
  const int row0 = blockIdx.x * 16;
  const int n0 = wave * 64;

  constexpr int KT = FF / 32;
  v8f acc[4] = {};
  v16bf a_cur = load_a_frag(xbf, FF, row0, 0);
  v16bf b_cur[4];
#pragma unroll
  for (int j = 0; j < 4; ++j) b_cur[j] = load_b_frag(fwT, FF, 16 * j + n0, 0);

#pragma unroll
  for (int ki = 0; ki < KT; ++ki) {
    v16bf a_nxt{};
    v16bf b_nxt[4];
    if (ki + 1 < KT) {
      const int kn = (ki + 1) * 32;
      a_nxt = load_a_frag(xbf, FF, row0, kn);
#pragma unroll
      for (int j = 0; j < 4; ++j) b_nxt[j] = load_b_frag(fwT, FF, 16 * j + n0, kn);
    }
#pragma unroll
    for (int j = 0; j < 4; ++j) acc[j] = wmma_bf16(a_cur, b_cur[j], acc[j]);
    if (ki + 1 < KT) {
      a_cur = a_nxt;
#pragma unroll
      for (int j = 0; j < 4; ++j) b_cur[j] = b_nxt[j];
    }
  }

  const int hi = lane >> 4, ln = lane & 15;
#pragma unroll
  for (int j = 0; j < 4; ++j)
#pragma unroll
    for (int r = 0; r < 8; ++r) {
      int rr = r + hi * 8;
      int cc = n0 + 16 * j + ln;
      zs[rr][cc] = cosf(acc[j][r] + fb[cc]);
    }
  __syncthreads();

  const int row = tid >> 4, seg = tid & 15;
  float s = 0.f, q = 0.f;
#pragma unroll 4
  for (int c = seg * 32; c < seg * 32 + 32; ++c) {
    float v = zs[row][c];
    s += v;
    q += v * v;
  }
  red_s[row][seg] = s;
  red_q[row][seg] = q;
  __syncthreads();

  float S = 0.f, Q = 0.f;
#pragma unroll
  for (int i = 0; i < 16; ++i) {
    S += red_s[row][i];
    Q += red_q[row][i];
  }
  const float mu = S * (1.0f / DD);
  const float var = Q * (1.0f / DD) - mu * mu;
  const float rs = rsqrtf(var + 1e-3f);

  for (int c = seg * 32; c < seg * 32 + 32; ++c) {
    float v = (zs[row][c] - mu) * rs * gamma[c] + beta[c];
    zbf[(size_t)(row0 + row) * DD + c] = (__bf16)v;
  }
}

// ---------------------------------------------------------------------------
// Gate pre-activations: a_g = z @ W_g + b_g   (stored time-major f32:
// a_ws[g][t][b][d]).  Grid (rows/32, D/64, 4 gates), wave -> one 16x16 tile.
// ---------------------------------------------------------------------------
__global__ __launch_bounds__(256) void k_gates(
    const __bf16* __restrict__ zbf, const __bf16* __restrict__ WT,
    const float* __restrict__ bi, const float* __restrict__ bff,
    const float* __restrict__ bo, const float* __restrict__ bc,
    float* __restrict__ a_ws) {
  const int g = blockIdx.z;
  const int row0 = blockIdx.x * 32, col0 = blockIdx.y * 64;
  const int tid = threadIdx.x, lane = tid & 31, wave = tid >> 5;
  const int m0 = row0 + (wave >> 2) * 16;
  const int n0 = col0 + (wave & 3) * 16;
  const __bf16* Bt = WT + (size_t)g * DD * DD;

  constexpr int KT = DD / 32;
  v8f acc = {};
  v16bf a_cur = load_a_frag(zbf, DD, m0, 0);
  v16bf b_cur = load_b_frag(Bt, DD, n0, 0);
#pragma unroll
  for (int ki = 0; ki < KT; ++ki) {
    v16bf a_nxt{}, b_nxt{};
    if (ki + 1 < KT) {
      const int kn = (ki + 1) * 32;
      if (kn + 32 < DD)   // warm the weight stream two K-blocks ahead
        __builtin_prefetch(Bt + (size_t)(n0 + (lane & 15)) * DD + kn + 32, 0, 0);
      a_nxt = load_a_frag(zbf, DD, m0, kn);
      b_nxt = load_b_frag(Bt, DD, n0, kn);
    }
    acc = wmma_bf16(a_cur, b_cur, acc);
    if (ki + 1 < KT) {
      a_cur = a_nxt;
      b_cur = b_nxt;
    }
  }

  const float* bias = (g == 0) ? bi : (g == 1) ? bff : (g == 2) ? bo : bc;
  const int hi = lane >> 4, ln = lane & 15;
  const int col = n0 + ln;
  const float bv = bias[col];
#pragma unroll
  for (int r = 0; r < 8; ++r) {
    int grow = m0 + r + hi * 8;        // = b*T + t
    int bb = grow >> 7;                // / T (T=128)
    int tt = grow & 127;
    size_t off = (((size_t)g * TT + tt) * BB + bb) * DD + col;
    a_ws[off] = acc[r] + bv;
  }
}

// ---------------------------------------------------------------------------
// One LSTM time step: pre_g = h_{t-1} @ R_g, then fused elementwise update.
// Grid: 8 blocks (64 cols each).  8 waves: wave = (gate<<1)|rowtile, each
// computing 16 rows x 64 cols of its gate.  Results staged in LDS.
// Double-buffered K-loop + prefetch of the recurrent weight stream.
// ---------------------------------------------------------------------------
__global__ __launch_bounds__(256) void k_step(
    const __bf16* __restrict__ hprev, int hstride,
    const __bf16* __restrict__ RT, const float* __restrict__ a_ws, int t,
    float* __restrict__ c_ws, __bf16* __restrict__ hall) {
  __shared__ float pre[4][BB][64];
  const int tid = threadIdx.x, lane = tid & 31, wave = tid >> 5;
  const int base = blockIdx.x * 64;
  const int gg = wave >> 1;
  const int m0 = (wave & 1) * 16;
  const __bf16* Bt = RT + (size_t)gg * DD * DD;

  constexpr int KT = DD / 32;
  v8f acc[4] = {};
  v16bf a_cur = load_a_frag(hprev, hstride, m0, 0);
  v16bf b_cur[4];
#pragma unroll
  for (int j = 0; j < 4; ++j) b_cur[j] = load_b_frag(Bt, DD, base + 16 * j, 0);

#pragma unroll
  for (int ki = 0; ki < KT; ++ki) {
    v16bf a_nxt{};
    v16bf b_nxt[4];
    if (ki + 1 < KT) {
      const int kn = (ki + 1) * 32;
      if (kn + 32 < DD)
        __builtin_prefetch(Bt + (size_t)(base + (lane & 15)) * DD + kn + 32, 0, 0);
      a_nxt = load_a_frag(hprev, hstride, m0, kn);
#pragma unroll
      for (int j = 0; j < 4; ++j) b_nxt[j] = load_b_frag(Bt, DD, base + 16 * j, kn);
    }
#pragma unroll
    for (int j = 0; j < 4; ++j) acc[j] = wmma_bf16(a_cur, b_cur[j], acc[j]);
    if (ki + 1 < KT) {
      a_cur = a_nxt;
#pragma unroll
      for (int j = 0; j < 4; ++j) b_cur[j] = b_nxt[j];
    }
  }

  const int hi = lane >> 4, ln = lane & 15;
#pragma unroll
  for (int j = 0; j < 4; ++j)
#pragma unroll
    for (int r = 0; r < 8; ++r)
      pre[gg][m0 + r + hi * 8][16 * j + ln] = acc[j][r];
  __syncthreads();

  const size_t gstride = (size_t)TT * BB * DD;
#pragma unroll
  for (int i = 0; i < 8; ++i) {
    int idx = tid * 8 + i;             // 0..2047 over 32 rows x 64 cols
    int bb = idx >> 6;
    int cc = idx & 63;
    int d = base + cc;
    size_t aoff = ((size_t)t * BB + bb) * DD + d;
    float zi = a_ws[aoff];
    float zf = a_ws[aoff + gstride];
    float zo = a_ws[aoff + 2 * gstride];
    float zc = a_ws[aoff + 3 * gstride];
    float ig = 1.f / (1.f + expf(-(zi + pre[0][bb][cc])));
    float fg = 1.f / (1.f + expf(-(zf + pre[1][bb][cc])));
    float og = sinf(zo + pre[2][bb][cc]);
    float gc = tanhf(zc + pre[3][bb][cc]);
    float cold = c_ws[bb * DD + d];
    float cnew = fg * cold + ig * gc;
    c_ws[bb * DD + d] = cnew;
    float h = og * cnew;
    hall[((size_t)bb * TT + t) * DD + d] = (__bf16)h;
  }
}

// ---------------------------------------------------------------------------
// Projection: out[b][o][t] = h_all @ proj_w + proj_b (output transposed)
// ---------------------------------------------------------------------------
__global__ __launch_bounds__(256) void k_proj(
    const __bf16* __restrict__ hall, const __bf16* __restrict__ pT,
    const float* __restrict__ pb, float* __restrict__ out) {
  const int row0 = blockIdx.x * 32, col0 = blockIdx.y * 64;
  const int tid = threadIdx.x, lane = tid & 31, wave = tid >> 5;
  const int m0 = row0 + (wave >> 2) * 16;
  const int n0 = col0 + (wave & 3) * 16;

  constexpr int KT = DD / 32;
  v8f acc = {};
  v16bf a_cur = load_a_frag(hall, DD, m0, 0);
  v16bf b_cur = load_b_frag(pT, DD, n0, 0);
#pragma unroll
  for (int ki = 0; ki < KT; ++ki) {
    v16bf a_nxt{}, b_nxt{};
    if (ki + 1 < KT) {
      const int kn = (ki + 1) * 32;
      a_nxt = load_a_frag(hall, DD, m0, kn);
      b_nxt = load_b_frag(pT, DD, n0, kn);
    }
    acc = wmma_bf16(a_cur, b_cur, acc);
    if (ki + 1 < KT) {
      a_cur = a_nxt;
      b_cur = b_nxt;
    }
  }

  const int hi = lane >> 4, ln = lane & 15;
  const int o = n0 + ln;
  const float bv = pb[o];
#pragma unroll
  for (int r = 0; r < 8; ++r) {
    int grow = m0 + r + hi * 8;        // = b*T + t
    int bb = grow >> 7;
    int tt = grow & 127;
    out[((size_t)bb * OO + o) * TT + tt] = acc[r] + bv;
  }
}

// ---------------------------------------------------------------------------
// Host-side orchestration
// ---------------------------------------------------------------------------
static inline size_t align256(size_t v) { return (v + 255) & ~(size_t)255; }

extern "C" void kernel_launch(void* const* d_in, const int* in_sizes, int n_in,
                              void* d_out, int out_size, void* d_ws, size_t ws_size,
                              hipStream_t stream) {
  (void)in_sizes; (void)n_in; (void)out_size; (void)ws_size;

  const float* x         = (const float*)d_in[0];
  const float* fourier_w = (const float*)d_in[1];
  const float* fourier_b = (const float*)d_in[2];
  const float* ln_gamma  = (const float*)d_in[3];
  const float* ln_beta   = (const float*)d_in[4];
  const float* Wg[4] = {(const float*)d_in[5], (const float*)d_in[6],
                        (const float*)d_in[7], (const float*)d_in[8]};
  const float* Rg[4] = {(const float*)d_in[9], (const float*)d_in[10],
                        (const float*)d_in[11], (const float*)d_in[12]};
  const float* bi = (const float*)d_in[13];
  const float* bf = (const float*)d_in[14];
  const float* bo = (const float*)d_in[15];
  const float* bc = (const float*)d_in[16];
  const float* init_h = (const float*)d_in[17];
  const float* init_c = (const float*)d_in[18];
  const float* proj_w = (const float*)d_in[19];
  const float* proj_b = (const float*)d_in[20];
  float* out = (float*)d_out;

  // Workspace carve-up
  uint8_t* p = (uint8_t*)d_ws;
  size_t off = 0;
  __bf16* xbf = (__bf16*)(p + off); off = align256(off + (size_t)BB*TT*FF*2);
  __bf16* fwT = (__bf16*)(p + off); off = align256(off + (size_t)FF*DD*2);
  __bf16* WT  = (__bf16*)(p + off); off = align256(off + (size_t)4*DD*DD*2);
  __bf16* RT  = (__bf16*)(p + off); off = align256(off + (size_t)4*DD*DD*2);
  __bf16* pT  = (__bf16*)(p + off); off = align256(off + (size_t)DD*OO*2);
  __bf16* zbf = (__bf16*)(p + off); off = align256(off + (size_t)BB*TT*DD*2);
  float*  a_ws = (float*)(p + off); off = align256(off + (size_t)4*TT*BB*DD*4);
  __bf16* hall = (__bf16*)(p + off); off = align256(off + (size_t)BB*TT*DD*2);
  __bf16* hinit = (__bf16*)(p + off); off = align256(off + (size_t)BB*DD*2);
  float*  c_ws = (float*)(p + off); off = align256(off + (size_t)BB*DD*4);

  // 1) pack inputs / weights to bf16 (weights transposed for B fragments)
  k_pack_x<<<(BB*FF*TT + 255)/256, 256, 0, stream>>>(x, xbf);
  k_pack_wT<<<(FF*DD + 255)/256, 256, 0, stream>>>(fourier_w, fwT, FF, DD);
  for (int g = 0; g < 4; ++g) {
    k_pack_wT<<<(DD*DD + 255)/256, 256, 0, stream>>>(Wg[g], WT + (size_t)g*DD*DD, DD, DD);
    k_pack_wT<<<(DD*DD + 255)/256, 256, 0, stream>>>(Rg[g], RT + (size_t)g*DD*DD, DD, DD);
  }
  k_pack_wT<<<(DD*OO + 255)/256, 256, 0, stream>>>(proj_w, pT, DD, OO);
  k_init_state<<<(BB*DD + 255)/256, 256, 0, stream>>>(init_h, init_c, hinit, c_ws);

  // 2) Fourier features + LayerNorm -> z (bf16)
  k_fourier_ln<<<(BB*TT)/16, 256, 0, stream>>>(xbf, fwT, fourier_b, ln_gamma,
                                               ln_beta, zbf);

  // 3) Input-side gate GEMMs (time-major f32 pre-activations)
  k_gates<<<dim3((BB*TT)/32, DD/64, 4), 256, 0, stream>>>(zbf, WT, bi, bf, bo,
                                                          bc, a_ws);

  // 4) Serial LSTM scan: 128 graph-captured step kernels
  for (int t = 0; t < TT; ++t) {
    const __bf16* hprev = (t == 0) ? hinit : (hall + (size_t)(t - 1) * DD);
    int hstride = (t == 0) ? DD : TT * DD;
    k_step<<<DD/64, 256, 0, stream>>>(hprev, hstride, RT, a_ws, t, c_ws, hall);
  }

  // 5) Projection (+ transposed store to [B, OUT, T])
  k_proj<<<dim3((BB*TT)/32, OO/64), 256, 0, stream>>>(hall, pT, proj_b, out);
}
